// Net_12524124635293
// MI455X (gfx1250) — compile-verified, matrix operand order checked by reading, not run on hardware
//
#include <hip/hip_runtime.h>
#include <math.h>

#define NNODES 8192
#define NEDGES 131072
#define DIN    64
#define HID    128
#define NHEADS 4
#define ADJW   (NNODES/32)   // 256 words of adjacency bits per row
#define ALPHA  0.2f
#define BNEPS  1e-5f
#define BPITCH 136           // 128 + 8 halves pad -> 272B row pitch, bank-conflict free
#define CEILDIV(a,b) (((a)+(b)-1)/(b))

typedef __attribute__((ext_vector_type(16))) _Float16 v16h;
typedef __attribute__((ext_vector_type(8)))  _Float16 v8h;
typedef __attribute__((ext_vector_type(8)))  float    v8f;

__device__ __forceinline__ float lrelu(float x){ return x > 0.f ? x : ALPHA * x; }
__device__ __forceinline__ v16h cat8(v8h lo, v8h hi){
  return __builtin_shufflevector(lo, hi, 0,1,2,3,4,5,6,7,8,9,10,11,12,13,14,15);
}

// ---------------------------------------------------------------------------
// Elementwise / conversion kernels
// ---------------------------------------------------------------------------
__global__ void k_bn_to_f16(const float* __restrict__ x, const float* __restrict__ g,
                            const float* __restrict__ b, _Float16* __restrict__ out,
                            int M, int N) {
  int t = blockIdx.x * blockDim.x + threadIdx.x;
  if (t >= M * N) return;
  int c = t % N;
  out[t] = (_Float16)(x[t] * (g[c] * rsqrtf(1.f + BNEPS)) + b[c]);
}

__global__ void k_transpose_h(const float* __restrict__ in, _Float16* __restrict__ out,
                              int K, int N) {  // in [K,N] f32 -> out [N,K] f16
  int t = blockIdx.x * blockDim.x + threadIdx.x;
  if (t >= K * N) return;
  int k = t / N, c = t % N;
  out[(size_t)c * K + k] = (_Float16)in[t];
}

__global__ void k_fill(float* p, float v, int n) {
  int t = blockIdx.x * blockDim.x + threadIdx.x;
  if (t < n) p[t] = v;
}

// ---------------------------------------------------------------------------
// Generic WMMA GEMM: C[M,N] = A[M,K] @ B[K,N]
//   A  : f16 row-major [M,K]
//   Bt : f16 row-major [N,K]  (B transposed so lane loads are contiguous in K)
//   mode 0: none       mode 2: (x + bias)*g/sqrt(1+eps) + beta, then ReLU
//   Cf (f32 [M,N]) / Ch (f16 [M,N]) / Cth (f16 [N,M]) written if non-null.
// One wave computes one 16x16 tile; K must be a multiple of 32.
// ---------------------------------------------------------------------------
__global__ void k_gemm(const _Float16* __restrict__ A, const _Float16* __restrict__ Bt,
                       float* Cf, _Float16* Ch, _Float16* Cth,
                       const float* bias, const float* g, const float* beta,
                       int M, int N, int K, int mode) {
  const int lane = threadIdx.x & 31;
  const int half = lane >> 4;
  const int idx  = lane & 15;
  const int m0 = blockIdx.x * 16;
  const int n0 = blockIdx.y * 16;
  const _Float16* arow = A  + (size_t)(m0 + idx) * K + half * 8;
  const _Float16* brow = Bt + (size_t)(n0 + idx) * K + half * 8;
  v8f acc = {};
  for (int k0 = 0; k0 < K; k0 += 32) {
    v8h alo = *(const v8h*)(arow + k0);
    v8h ahi = *(const v8h*)(arow + k0 + 16);
    v8h blo = *(const v8h*)(brow + k0);
    v8h bhi = *(const v8h*)(brow + k0 + 16);
    v16h a = cat8(alo, ahi);
    v16h b = cat8(blo, bhi);
    acc = __builtin_amdgcn_wmma_f32_16x16x32_f16(false, a, false, b, (short)0, acc,
                                                 false, false);
  }
  const int n = n0 + idx;
  float bi = bias ? bias[n] : 0.f;
  float sc = 1.f, sh = 0.f;
  if (mode == 2) { sc = g[n] * rsqrtf(1.f + BNEPS); sh = beta[n]; }
#pragma unroll
  for (int v = 0; v < 8; ++v) {
    int m = m0 + half * 8 + v;               // D layout: VGPR v -> row v + 8*half
    float val = acc[v] + bi;
    if (mode == 2) val = fmaxf(val * sc + sh, 0.f);
    if (Cf)  Cf [(size_t)m * N + n] = val;
    if (Ch)  Ch [(size_t)m * N + n] = (_Float16)val;
    if (Cth) Cth[(size_t)n * M + m] = (_Float16)val;
  }
}

// ---------------------------------------------------------------------------
// GAT attention
// ---------------------------------------------------------------------------
__global__ void k_build_adj(const int* __restrict__ src, const int* __restrict__ dst,
                            unsigned* __restrict__ adj, int E) {
  int e = blockIdx.x * blockDim.x + threadIdx.x;
  if (e >= E) return;
  int s = src[e], d = dst[e];
  atomicOr(&adj[(size_t)s * ADJW + (d >> 5)], 1u << (d & 31));
  atomicOr(&adj[(size_t)d * ADJW + (s >> 5)], 1u << (s & 31));
}

__global__ void k_rowdot(const float* __restrict__ Wh, const float* __restrict__ a,
                         float* __restrict__ f1, float* __restrict__ f2, int Nn, int F) {
  int i = blockIdx.x * blockDim.x + threadIdx.x;
  if (i >= Nn) return;
  float s1 = 0.f, s2 = 0.f;
  for (int c = 0; c < F; ++c) {
    float w = Wh[(size_t)i * F + c];
    s1 += w * a[c];
    s2 += w * a[F + c];
  }
  f1[i] = s1; f2[i] = s2;
}

// One wave per row: emax = lrelu(max(f1[i]+f2[j])) over adjacent j (lrelu monotone),
// sinv = 1/sum(exp(e - emax)); isolated rows -> uniform 1/N attention.
__global__ void k_att_stats(const unsigned* __restrict__ adj, const float* __restrict__ f1,
                            const float* __restrict__ f2, float* __restrict__ emax,
                            float* __restrict__ sinv, int* __restrict__ iso, int Nn) {
  int i = blockIdx.x;
  int lane = threadIdx.x;
  int words = Nn >> 5;
  float f1i = f1[i];
  const unsigned* arow = adj + (size_t)i * words;
  float mraw = -3.0e38f;
  int cnt = 0;
  for (int w = lane; w < words; w += 32) {
    unsigned bits = arow[w];
    while (bits) {
      int b = __ffs(bits) - 1;
      bits &= bits - 1;
      mraw = fmaxf(mraw, f1i + f2[w * 32 + b]);
      cnt++;
    }
  }
  for (int o = 16; o; o >>= 1) {
    mraw = fmaxf(mraw, __shfl_xor(mraw, o, 32));
    cnt += __shfl_xor(cnt, o, 32);
  }
  float em = lrelu(mraw);
  float s = 0.f;
  for (int w = lane; w < words; w += 32) {
    unsigned bits = arow[w];
    while (bits) {
      int b = __ffs(bits) - 1;
      bits &= bits - 1;
      s += __expf(lrelu(f1i + f2[w * 32 + b]) - em);
    }
  }
  for (int o = 16; o; o >>= 1) s += __shfl_xor(s, o, 32);
  if (lane == 0) {
    if (cnt == 0) { emax[i] = 0.f; sinv[i] = 1.f / (float)Nn; iso[i] = 1; }
    else          { emax[i] = em;  sinv[i] = 1.f / s;         iso[i] = 0; }
  }
}

// ---------------------------------------------------------------------------
// Flash-style attention aggregation, LDS-blocked:
//   out[i,:] = (1/s_i) * sum_j exp(e_ij - m_i) * Wh[j,:]
// Block = 8 waves = 128 output rows x 128 cols. A 128-j B-panel of WhT is staged
// into LDS with gfx1250 async copies (global_load_async_to_lds_b128, ASYNCcnt),
// giving 8x operand reuse (128 FLOP per L2 byte instead of 16).
// A-fragment (16x32 f16) built on the fly per the CDNA5 layout:
//   lane = 16*half + m ; element e holds K-offset (e&8?16:0) + 8*half + (e&7)
// LDS panel pitch = 136 halves (272 B = 68 dwords) -> the 16 fragment rows hit
// all 64 banks exactly once per ds_load_b128 (conflict-free).
// ---------------------------------------------------------------------------
__global__ void __launch_bounds__(256)
k_att_agg(const unsigned* __restrict__ adj, const float* __restrict__ f1,
          const float* __restrict__ f2, const float* __restrict__ emax,
          const float* __restrict__ sinv, const int* __restrict__ iso,
          const _Float16* __restrict__ WhT, float* __restrict__ out,
          int Nn, int F) {
  __shared__ _Float16 panelB[128 * BPITCH];  // [col 0..127][j-offset 0..127] padded
  __shared__ float    f2s[128];

  const int tid  = threadIdx.x;
  const int lane = tid & 31;
  const int wv   = tid >> 5;                 // wave 0..7 -> 16-row strip
  const int half = lane >> 4;
  const int idx  = lane & 15;
  const int i0   = blockIdx.x * 128 + wv * 16;
  const int r    = i0 + idx;                 // row this lane's A-frag belongs to
  const float f1r  = f1[r];
  const float mr   = emax[r];
  const float isoR = iso[r] ? 1.f : 0.f;
  const unsigned* arow = adj + (size_t)r * (Nn >> 5);

  v8f zero = {};
  v8f acc[8];
#pragma unroll
  for (int c = 0; c < 8; ++c) acc[c] = zero;

  for (int j0 = 0; j0 < Nn; j0 += 128) {
    // stage f2 slice (plain LDS store) + 32 KB B-panel via async copy to LDS
    if (tid < 128) f2s[tid] = f2[j0 + tid];
#pragma unroll
    for (int q = 0; q < 8; ++q) {
      int ch   = tid + q * 256;              // 2048 16B chunks total
      int c    = ch >> 4;                    // panel column (0..127)
      int part = ch & 15;                    // 16B chunk within the 256B row
      unsigned lofs = (unsigned)(size_t)&panelB[c * BPITCH + part * 8];
      unsigned long long ga =
          (unsigned long long)(size_t)(WhT + (size_t)c * Nn + j0 + part * 8);
      asm volatile("global_load_async_to_lds_b128 %0, %1, off"
                   :: "v"(lofs), "v"(ga) : "memory");
    }
    asm volatile("s_wait_asynccnt 0x0" ::: "memory");
    __syncthreads();

#pragma unroll
    for (int jj0 = 0; jj0 < 128; jj0 += 32) {
      const int jb = j0 + jj0;               // 32-aligned -> one adjacency word
      unsigned wbits = arow[jb >> 5];
      v16h a;
#pragma unroll
      for (int e = 0; e < 16; ++e) {
        int off = ((e & 8) ? 16 : 0) + half * 8 + (e & 7);  // 0..31
        float p;
        if ((wbits >> off) & 1u)
          p = __expf(lrelu(f1r + f2s[jj0 + off]) - mr);
        else
          p = isoR;                          // isolated row -> uniform weights
        a[e] = (_Float16)p;
      }
#pragma unroll
      for (int c = 0; c < 8; ++c) {
        const _Float16* wt = &panelB[(c * 16 + idx) * BPITCH + jj0 + half * 8];
        v8h blo = *(const v8h*)(wt);
        v8h bhi = *(const v8h*)(wt + 16);
        v16h b = cat8(blo, bhi);
        acc[c] = __builtin_amdgcn_wmma_f32_16x16x32_f16(false, a, false, b, (short)0,
                                                        acc[c], false, false);
      }
    }
    __syncthreads();                         // protect panel before next fill
  }
#pragma unroll
  for (int c = 0; c < 8; ++c) {
#pragma unroll
    for (int v = 0; v < 8; ++v) {
      int m = i0 + half * 8 + v;
      out[(size_t)m * F + (c * 16 + idx)] = acc[c][v] * sinv[m];
    }
  }
}

__global__ void k_elu_to_cat(const float* __restrict__ in, _Float16* __restrict__ hcat,
                             int Nn, int coff, int totC) {
  int t = blockIdx.x * blockDim.x + threadIdx.x;
  if (t >= Nn * HID) return;
  int r = t >> 7, c = t & 127;
  float x = in[t];
  float y = x > 0.f ? x : (__expf(x) - 1.f);
  hcat[(size_t)r * totC + coff + c] = (_Float16)y;
}

__global__ void k_relu_logsoftmax(const float* __restrict__ in, float* __restrict__ outf,
                                  _Float16* __restrict__ outh, int F) {
  int i = blockIdx.x;
  int lane = threadIdx.x;
  float v[4];
  float mx = -3.0e38f;
#pragma unroll
  for (int q = 0; q < 4; ++q) {
    float x = fmaxf(in[(size_t)i * F + q * 32 + lane], 0.f);
    v[q] = x; mx = fmaxf(mx, x);
  }
  for (int o = 16; o; o >>= 1) mx = fmaxf(mx, __shfl_xor(mx, o, 32));
  float s = 0.f;
#pragma unroll
  for (int q = 0; q < 4; ++q) s += __expf(v[q] - mx);
  for (int o = 16; o; o >>= 1) s += __shfl_xor(s, o, 32);
  float lse = mx + logf(s);
#pragma unroll
  for (int q = 0; q < 4; ++q) {
    float y = v[q] - lse;
    outf[(size_t)i * F + q * 32 + lane] = y;
    outh[(size_t)i * F + q * 32 + lane] = (_Float16)y;
  }
}

// ---------------------------------------------------------------------------
// GCN + TopK pooling
// ---------------------------------------------------------------------------
__global__ void k_deg(const int* __restrict__ dst, const float* __restrict__ ew,
                      float* __restrict__ deg, int E) {
  int e = blockIdx.x * blockDim.x + threadIdx.x;
  if (e >= E) return;
  atomicAdd(&deg[dst[e]], ew[e]);
}

__global__ void k_dinv(const float* __restrict__ deg, float* __restrict__ dinv, int n) {
  int i = blockIdx.x * blockDim.x + threadIdx.x;
  if (i >= n) return;
  dinv[i] = rsqrtf(deg[i] + 1.0f);   // +1 self loop; always > 0
}

__global__ void k_gcn_scatter(const int* __restrict__ src, const int* __restrict__ dst,
                              const float* __restrict__ ew, const float* __restrict__ dinv,
                              const float* __restrict__ h, float* __restrict__ out,
                              int E, int dout, int shift) {
  int t = blockIdx.x * blockDim.x + threadIdx.x;
  if (t >= E * dout) return;
  int e = t >> shift;
  int c = t & (dout - 1);
  float w = ew[e];
  if (w == 0.f) return;
  int s = src[e], d = dst[e];
  float norm = dinv[s] * w * dinv[d];
  atomicAdd(&out[(size_t)d * dout + c], norm * h[(size_t)s * dout + c]);
}

__global__ void k_gcn_fin(float* __restrict__ agg, const float* __restrict__ h,
                          const float* __restrict__ dinv, const float* __restrict__ b,
                          int n, int dout) {
  int t = blockIdx.x * blockDim.x + threadIdx.x;
  if (t >= n * dout) return;
  int i = t / dout, c = t % dout;
  float dv = dinv[i];
  agg[t] = fmaxf(agg[t] + dv * dv * h[t] + b[c], 0.f);
}

__global__ void k_score(const float* __restrict__ x, const float* __restrict__ w,
                        float* __restrict__ score, int n, int d) {
  int i = blockIdx.x * blockDim.x + threadIdx.x;
  if (i >= n) return;
  float nrm = 0.f, dp = 0.f;
  for (int c = 0; c < d; ++c) nrm += w[c] * w[c];
  nrm = sqrtf(nrm);
  for (int c = 0; c < d; ++c) dp += x[(size_t)i * d + c] * w[c];
  score[i] = tanhf(dp / nrm);
}

// Exact top-k via rank (stable, descending, ties by index — matches lax.top_k).
__global__ void k_rank(const float* __restrict__ score, const float* __restrict__ x,
                       _Float16* __restrict__ xnewh, int* __restrict__ map,
                       int n, int k, int din) {
  int i = blockIdx.x * blockDim.x + threadIdx.x;
  if (i >= n) return;
  float si = score[i];
  int r = 0;
  for (int j = 0; j < n; ++j) {
    float sj = score[j];
    r += (sj > si) || (sj == si && j < i);
  }
  if (r < k) {
    map[i] = r;
    for (int c = 0; c < din; ++c)
      xnewh[(size_t)r * din + c] = (_Float16)(x[(size_t)i * din + c] * si);
  } else {
    map[i] = -1;
  }
}

__global__ void k_edge_remap(const int* __restrict__ src, const int* __restrict__ dst,
                             const float* __restrict__ ew, const int* __restrict__ map,
                             int* __restrict__ ns, int* __restrict__ nd,
                             float* __restrict__ nw, int E) {
  int e = blockIdx.x * blockDim.x + threadIdx.x;
  if (e >= E) return;
  int a = map[src[e]], b = map[dst[e]];
  bool valid = (a >= 0) && (b >= 0);
  ns[e] = valid ? a : 0;
  nd[e] = valid ? b : 0;
  nw[e] = valid ? ew[e] : 0.f;
}

__global__ void k_final(const float* __restrict__ x, const float* __restrict__ W,
                        const float* __restrict__ b, float* __restrict__ out,
                        int n, int d) {
  __shared__ float col[16];
  int t = threadIdx.x;
  if (t < d) {
    float s = 0.f;
    for (int r = 0; r < n; ++r) s += x[(size_t)r * d + t];
    col[t] = s / (float)n;
  }
  __syncthreads();
  if (t == 0) {
    float acc = b[0];
    for (int c = 0; c < d; ++c) acc += col[c] * W[c];
    out[0] = acc * 100.f;
  }
}

// ---------------------------------------------------------------------------
// Host launcher
// ---------------------------------------------------------------------------
extern "C" void kernel_launch(void* const* d_in, const int* in_sizes, int n_in,
                              void* d_out, int out_size, void* d_ws, size_t ws_size,
                              hipStream_t stream) {
  (void)in_sizes; (void)n_in; (void)out_size; (void)ws_size;
  // pytree flatten order: top-level dict insertion order (x, params, edge_index,
  // batch); params dict sorted by key: bn_in{b,g}, conv[{W,b}x4], gat_att[{W,a}x4],
  // gat_out{W,a}, lin{W,b}, mlp[{W,b,beta,g}x2], pool[3]
  const float* x     = (const float*)d_in[0];
  const float* bn_b  = (const float*)d_in[1];
  const float* bn_g  = (const float*)d_in[2];
  const float* convW[4] = {(const float*)d_in[3], (const float*)d_in[5],
                           (const float*)d_in[7], (const float*)d_in[9]};
  const float* convB[4] = {(const float*)d_in[4], (const float*)d_in[6],
                           (const float*)d_in[8], (const float*)d_in[10]};
  const float* gatW[4]  = {(const float*)d_in[11], (const float*)d_in[13],
                           (const float*)d_in[15], (const float*)d_in[17]};
  const float* gatA[4]  = {(const float*)d_in[12], (const float*)d_in[14],
                           (const float*)d_in[16], (const float*)d_in[18]};
  const float* outW  = (const float*)d_in[19];
  const float* outA  = (const float*)d_in[20];
  const float* linW  = (const float*)d_in[21];
  const float* linB  = (const float*)d_in[22];
  const float* mlpW[2]    = {(const float*)d_in[23], (const float*)d_in[27]};
  const float* mlpB[2]    = {(const float*)d_in[24], (const float*)d_in[28]};
  const float* mlpBeta[2] = {(const float*)d_in[25], (const float*)d_in[29]};
  const float* mlpG[2]    = {(const float*)d_in[26], (const float*)d_in[30]};
  const float* poolW[3]   = {(const float*)d_in[31], (const float*)d_in[32],
                             (const float*)d_in[33]};
  const int* ei   = (const int*)d_in[34];
  const int* src0 = ei;
  const int* dst0 = ei + NEDGES;

  // deterministic bump allocator over d_ws
  char* p = (char*)d_ws;
  auto alloc = [&](size_t bytes) -> void* {
    void* r = (void*)p;
    p += (bytes + 255) & ~(size_t)255;
    return r;
  };
  _Float16* xbf   = (_Float16*)alloc((size_t)NNODES * DIN * 2);
  _Float16* wT    = (_Float16*)alloc((size_t)512 * 128 * 2);      // reused weight^T
  _Float16* h1h   = (_Float16*)alloc((size_t)NNODES * HID * 2);
  _Float16* h2h   = (_Float16*)alloc((size_t)NNODES * HID * 2);
  unsigned* adj   = (unsigned*)alloc((size_t)NNODES * ADJW * 4);  // 8 MB bit matrix
  float*    Whf   = (float*)alloc((size_t)NNODES * HID * 4);
  _Float16* WhTh  = (_Float16*)alloc((size_t)NNODES * HID * 2);
  float* f1b   = (float*)alloc((size_t)NNODES * 4);
  float* f2b   = (float*)alloc((size_t)NNODES * 4);
  float* emaxb = (float*)alloc((size_t)NNODES * 4);
  float* sinvb = (float*)alloc((size_t)NNODES * 4);
  int*   isob  = (int*)alloc((size_t)NNODES * 4);
  float* attout = (float*)alloc((size_t)NNODES * HID * 4);
  _Float16* hcath = (_Float16*)alloc((size_t)NNODES * HID * NHEADS * 2);
  float*    gF = (float*)alloc((size_t)NNODES * HID * 4);
  _Float16* gH = (_Float16*)alloc((size_t)NNODES * HID * 2);
  float* hbuf   = (float*)alloc((size_t)NNODES * HID * 4);
  float* aggbuf = (float*)alloc((size_t)NNODES * HID * 4);
  float* degb   = (float*)alloc((size_t)NNODES * 4);
  float* dinvb  = (float*)alloc((size_t)NNODES * 4);
  float* scoreb = (float*)alloc((size_t)NNODES * 4);
  int*   mapb   = (int*)alloc((size_t)NNODES * 4);
  _Float16* xs1h = (_Float16*)alloc((size_t)4096 * 128 * 2);
  _Float16* xs2h = (_Float16*)alloc((size_t)2048 * 64 * 2);
  _Float16* xs3h = (_Float16*)alloc((size_t)1024 * 32 * 2);
  float* ew0 = (float*)alloc((size_t)NEDGES * 4);
  int* es1 = (int*)alloc((size_t)NEDGES * 4);
  int* ed1 = (int*)alloc((size_t)NEDGES * 4);
  float* ewn1 = (float*)alloc((size_t)NEDGES * 4);
  int* es2 = (int*)alloc((size_t)NEDGES * 4);
  int* ed2 = (int*)alloc((size_t)NEDGES * 4);
  float* ewn2 = (float*)alloc((size_t)NEDGES * 4);
  int* es3 = (int*)alloc((size_t)NEDGES * 4);
  int* ed3 = (int*)alloc((size_t)NEDGES * 4);
  float* ewn3 = (float*)alloc((size_t)NEDGES * 4);

  const int T = 256;

  // ---- input BN + NodeMLP (fused BN+ReLU GEMM epilogue) ----
  k_bn_to_f16<<<CEILDIV(NNODES * DIN, T), T, 0, stream>>>(x, bn_g, bn_b, xbf, NNODES, DIN);
  k_transpose_h<<<CEILDIV(DIN * HID, T), T, 0, stream>>>(mlpW[0], wT, DIN, HID);
  k_gemm<<<dim3(NNODES / 16, HID / 16), 32, 0, stream>>>(
      xbf, wT, nullptr, h1h, nullptr, mlpB[0], mlpG[0], mlpBeta[0], NNODES, HID, DIN, 2);
  k_transpose_h<<<CEILDIV(HID * HID, T), T, 0, stream>>>(mlpW[1], wT, HID, HID);
  k_gemm<<<dim3(NNODES / 16, HID / 16), 32, 0, stream>>>(
      h1h, wT, nullptr, h2h, nullptr, mlpB[1], mlpG[1], mlpBeta[1], NNODES, HID, HID, 2);

  // ---- adjacency bit matrix (8 MB, L2-resident) ----
  hipMemsetAsync(adj, 0, (size_t)NNODES * ADJW * 4, stream);
  k_build_adj<<<CEILDIV(NEDGES, T), T, 0, stream>>>(src0, dst0, adj, NEDGES);

  // ---- GAT heads (elu, concat) ----
  for (int hh = 0; hh < NHEADS; ++hh) {
    k_transpose_h<<<CEILDIV(HID * HID, T), T, 0, stream>>>(gatW[hh], wT, HID, HID);
    k_gemm<<<dim3(NNODES / 16, HID / 16), 32, 0, stream>>>(
        h2h, wT, Whf, nullptr, WhTh, nullptr, nullptr, nullptr, NNODES, HID, HID, 0);
    k_rowdot<<<CEILDIV(NNODES, T), T, 0, stream>>>(Whf, gatA[hh], f1b, f2b, NNODES, HID);
    k_att_stats<<<NNODES, 32, 0, stream>>>(adj, f1b, f2b, emaxb, sinvb, isob, NNODES);
    k_att_agg<<<NNODES / 128, 256, 0, stream>>>(adj, f1b, f2b, emaxb, sinvb, isob, WhTh,
                                                attout, NNODES, HID);
    k_elu_to_cat<<<CEILDIV(NNODES * HID, T), T, 0, stream>>>(attout, hcath, NNODES,
                                                             hh * HID, HID * NHEADS);
  }

  // ---- out attention layer, then ReLU + log_softmax ----
  k_transpose_h<<<CEILDIV(HID * NHEADS * HID, T), T, 0, stream>>>(outW, wT, HID * NHEADS, HID);
  k_gemm<<<dim3(NNODES / 16, HID / 16), 32, 0, stream>>>(
      hcath, wT, Whf, nullptr, WhTh, nullptr, nullptr, nullptr, NNODES, HID, HID * NHEADS, 0);
  k_rowdot<<<CEILDIV(NNODES, T), T, 0, stream>>>(Whf, outA, f1b, f2b, NNODES, HID);
  k_att_stats<<<NNODES, 32, 0, stream>>>(adj, f1b, f2b, emaxb, sinvb, isob, NNODES);
  k_att_agg<<<NNODES / 128, 256, 0, stream>>>(adj, f1b, f2b, emaxb, sinvb, isob, WhTh,
                                              attout, NNODES, HID);
  k_relu_logsoftmax<<<NNODES, 32, 0, stream>>>(attout, gF, gH, HID);

  // ---- GCN + TopK pipeline ----
  k_fill<<<CEILDIV(NEDGES, T), T, 0, stream>>>(ew0, 1.0f, NEDGES);

  auto gcn_stage = [&](const _Float16* Ah, int n, int din, int dout,
                       const float* W, const float* B,
                       const int* s, const int* d, const float* w) {
    k_transpose_h<<<CEILDIV(din * dout, T), T, 0, stream>>>(W, wT, din, dout);
    k_gemm<<<dim3(n / 16, dout / 16), 32, 0, stream>>>(
        Ah, wT, hbuf, nullptr, nullptr, nullptr, nullptr, nullptr, n, dout, din, 0);
    hipMemsetAsync(degb, 0, (size_t)n * 4, stream);
    hipMemsetAsync(aggbuf, 0, (size_t)n * dout * 4, stream);
    k_deg<<<CEILDIV(NEDGES, T), T, 0, stream>>>(d, w, degb, NEDGES);
    k_dinv<<<CEILDIV(n, T), T, 0, stream>>>(degb, dinvb, n);
    int shift = (dout == 128) ? 7 : (dout == 64) ? 6 : (dout == 32) ? 5 : 4;
    k_gcn_scatter<<<CEILDIV(NEDGES * dout, T), T, 0, stream>>>(s, d, w, dinvb, hbuf,
                                                               aggbuf, NEDGES, dout, shift);
    k_gcn_fin<<<CEILDIV(n * dout, T), T, 0, stream>>>(aggbuf, hbuf, dinvb, B, n, dout);
  };
  auto pool_stage = [&](int n, int din, int k, const float* pw, _Float16* xnh,
                        const int* s, const int* d, const float* w,
                        int* ns_, int* nd_, float* nw_) {
    k_score<<<CEILDIV(n, T), T, 0, stream>>>(aggbuf, pw, scoreb, n, din);
    k_rank<<<CEILDIV(n, T), T, 0, stream>>>(scoreb, aggbuf, xnh, mapb, n, k, din);
    k_edge_remap<<<CEILDIV(NEDGES, T), T, 0, stream>>>(s, d, w, mapb, ns_, nd_, nw_, NEDGES);
  };

  gcn_stage(gH, 8192, 128, 128, convW[0], convB[0], src0, dst0, ew0);
  pool_stage(8192, 128, 4096, poolW[0], xs1h, src0, dst0, ew0, es1, ed1, ewn1);
  gcn_stage(xs1h, 4096, 128, 64, convW[1], convB[1], es1, ed1, ewn1);
  pool_stage(4096, 64, 2048, poolW[1], xs2h, es1, ed1, ewn1, es2, ed2, ewn2);
  gcn_stage(xs2h, 2048, 64, 32, convW[2], convB[2], es2, ed2, ewn2);
  pool_stage(2048, 32, 1024, poolW[2], xs3h, es2, ed2, ewn2, es3, ed3, ewn3);
  gcn_stage(xs3h, 1024, 32, 16, convW[3], convB[3], es3, ed3, ewn3);

  // ---- global mean pool + final linear, x100 ----
  k_final<<<1, 32, 0, stream>>>(aggbuf, linW, linB, (float*)d_out, 1024, 16);
}
